// OrderedGNN_38019050504554
// MI455X (gfx1250) — matrix operand order repaired and checked by reading, not compiled, not run on hardware
//
#include <hip/hip_runtime.h>
#include <hip/hip_bf16.h>

// ---------------------------------------------------------------------------
// OrderedGNN for MI455X (gfx1250, wave32, WMMA).
// All dense algebra runs through v_wmma_f32_16x16x32_f16 (f16 in, f32 accum).
//
// Key restructure vs naive mapping: the gated aggregation is linear in the
// projection weights, so we aggregate h first and project the aggregates:
//   aggr_fwd = W_fwd (Σ_src α h_src),  aggr_bwd = W_bwd (Σ h_src − Σ α h_src)
// This halves the per-edge gather (one f16 row instead of two f32 rows) and
// lets the fwd/bwd/self projections write f16 straight into the [N,384]
// concat operand of the comb GEMM.
// ---------------------------------------------------------------------------

typedef _Float16 v4h  __attribute__((ext_vector_type(4)));
typedef _Float16 v8h  __attribute__((ext_vector_type(8)));
typedef _Float16 v16h __attribute__((ext_vector_type(16)));
typedef float    v8f  __attribute__((ext_vector_type(8)));

#define DCONST 128

// ---- WMMA operand loaders (contiguous k order, same for A and B) ----------------------
// A tile 16x32 (MxK), f16: lane L holds row m0+(L%16); halves [0:16) = K
// k0+16*(L/16)+0..15.  Two contiguous 16B loads.
__device__ __forceinline__ v16h load_a_tile(const _Float16* __restrict__ A,
                                            int m0, int lda, int k0, int lane) {
  int m    = m0 + (lane & 15);
  int khav = 16 * (lane >> 4);
  const _Float16* p = A + (size_t)m * lda + k0 + khav;
  v8h lo = *(const v8h*)(p);
  v8h hi = *(const v8h*)(p + 8);
  return __builtin_shufflevector(lo, hi, 0,1,2,3,4,5,6,7,8,9,10,11,12,13,14,15);
}

// B tile 32x16 (KxN) where B[k,n] = W[n,k] (row-major W over K): lane L holds
// W row n0+(L%16), 16 contiguous halves starting at k0+16*(L/16). One 32B load.
__device__ __forceinline__ v16h load_b_tile(const _Float16* __restrict__ W,
                                            int n0, int ldw, int k0, int lane) {
  int n    = n0 + (lane & 15);
  int khav = 16 * (lane >> 4);
  return *(const v16h*)(W + (size_t)n * ldw + k0 + khav);
}

// ---- Generic C = relu?(A[N,K] @ W[16*NT,K]^T + bias) -----------------------
// One wave per 16-row M tile, owning all NT column tiles. Output written at
// C[row*ldo + colBase + col] so projections can target slices of the concat
// buffer directly (f16) and/or an f32 buffer.
template <int NT, bool RELU>
__global__ void gemm_f16_kernel(const _Float16* __restrict__ A,
                                const _Float16* __restrict__ W,
                                const float*    __restrict__ bias,   // may be null
                                float*          __restrict__ Cf,     // may be null
                                _Float16*       __restrict__ Ch,     // may be null
                                int N, int K, int ldo, int colBase) {
  int wave = blockIdx.x * (blockDim.x >> 5) + (threadIdx.x >> 5);
  int lane = threadIdx.x & 31;
  int mtiles = N >> 4;
  if (wave >= mtiles) return;                // uniform per-wave: EXEC stays full
  int m0 = wave << 4;

  v8f acc[NT];
  const v8f vzero = {0.f,0.f,0.f,0.f,0.f,0.f,0.f,0.f};
#pragma unroll
  for (int t = 0; t < NT; ++t) acc[t] = vzero;

  for (int k0 = 0; k0 < K; k0 += 32) {
    v16h a = load_a_tile(A, m0, K, k0, lane);
#pragma unroll
    for (int t = 0; t < NT; ++t) {
      v16h b = load_b_tile(W, t * 16, K, k0, lane);
      acc[t] = __builtin_amdgcn_wmma_f32_16x16x32_f16(
          false, a, false, b, (short)0, acc[t], false, false);
    }
  }

  int colb = lane & 15;
  int half = lane >> 4;
#pragma unroll
  for (int t = 0; t < NT; ++t) {
    int col = t * 16 + colb;
    float bv = bias ? bias[col] : 0.f;
#pragma unroll
    for (int r = 0; r < 8; ++r) {
      int row = m0 + r + 8 * half;
      float v = acc[t][r] + bv;
      if (RELU) v = fmaxf(v, 0.f);
      size_t idx = (size_t)row * ldo + colBase + col;
      if (Cf) Cf[idx] = v;
      if (Ch) Ch[idx] = (_Float16)v;
    }
  }
}

// ---- Comb GEMM + bias + LayerNorm + ReLU + residual, in-place h update -----
// z = Zc[N,384] @ Wc[128,384]^T + bcomb ; LN over 128 cols ; h = relu(z)+h.
// One wave holds a full 16x128 row block in registers -> LN via shfl_xor only.
__global__ void comb_ln_kernel(const _Float16* __restrict__ Zc,
                               const _Float16* __restrict__ Wc,
                               const float* __restrict__ bcomb,
                               const float* __restrict__ gamma,
                               const float* __restrict__ beta,
                               float* __restrict__ hf,      // in: identity, out: new h
                               _Float16* __restrict__ hh,   // out: f16 mirror
                               int N) {
  const int K = 3 * DCONST;
  int wave = blockIdx.x * (blockDim.x >> 5) + (threadIdx.x >> 5);
  int lane = threadIdx.x & 31;
  int mtiles = N >> 4;
  if (wave >= mtiles) return;
  int m0 = wave << 4;

  v8f acc[8];
  const v8f vzero = {0.f,0.f,0.f,0.f,0.f,0.f,0.f,0.f};
#pragma unroll
  for (int t = 0; t < 8; ++t) acc[t] = vzero;

  for (int k0 = 0; k0 < K; k0 += 32) {
    v16h a = load_a_tile(Zc, m0, K, k0, lane);
#pragma unroll
    for (int t = 0; t < 8; ++t) {
      v16h b = load_b_tile(Wc, t * 16, K, k0, lane);
      acc[t] = __builtin_amdgcn_wmma_f32_16x16x32_f16(
          false, a, false, b, (short)0, acc[t], false, false);
    }
  }

  int colb = lane & 15;
  int half = lane >> 4;
  // bias
#pragma unroll
  for (int t = 0; t < 8; ++t) {
    float bv = bcomb[t * 16 + colb];
#pragma unroll
    for (int r = 0; r < 8; ++r) acc[t][r] += bv;
  }
  // Row statistics. Row (r + 8*half) lives entirely in this lane's 16-lane
  // half-group, cols spread across lanes; xor masks 1..8 stay within the half.
  float mu[8], rs[8];
#pragma unroll
  for (int r = 0; r < 8; ++r) {
    float sv = 0.f, qv = 0.f;
#pragma unroll
    for (int t = 0; t < 8; ++t) { float v = acc[t][r]; sv += v; qv += v * v; }
    for (int m = 1; m < 16; m <<= 1) {
      sv += __shfl_xor(sv, m, 32);
      qv += __shfl_xor(qv, m, 32);
    }
    float m1  = sv * (1.f / 128.f);
    float var = qv * (1.f / 128.f) - m1 * m1;
    mu[r] = m1;
    rs[r] = rsqrtf(var + 1e-5f);
  }
  // normalize, affine, relu, residual (in-place: each thread touches its own
  // elements only, read-then-write per element is race-free).
#pragma unroll
  for (int t = 0; t < 8; ++t) {
    int col = t * 16 + colb;
    float g = gamma[col], be = beta[col];
#pragma unroll
    for (int r = 0; r < 8; ++r) {
      int row = m0 + r + 8 * half;
      size_t idx = (size_t)row * DCONST + col;
      float v = (acc[t][r] - mu[r]) * rs[r] * g + be;
      v = fmaxf(v, 0.f) + hf[idx];
      hf[idx] = v;
      hh[idx] = (_Float16)v;
    }
  }
}

// ---- Edge phase: gather h[src] (f16), scatter α·h and h into f32 aggregates.
// One wave per edge; lane owns 4 columns (8B f16 gather, 2x4 f32 atomics).
__global__ void edge_aggr_kernel(const long long* __restrict__ ei,   // [2,E]
                                 const float* __restrict__ scores,   // [N,16] col0
                                 const _Float16* __restrict__ h16,
                                 float* __restrict__ agg_a,  // Σ α h_src   per dst
                                 float* __restrict__ agg_s,  // Σ h_src     per dst
                                 int E) {
  int gid  = blockIdx.x * blockDim.x + threadIdx.x;
  int e    = gid >> 5;
  int lane = gid & 31;
  if (e >= E) return;
  int s = (int)ei[e];
  int d = (int)ei[(size_t)E + e];
  float diff  = scores[(size_t)s * 16] - scores[(size_t)d * 16];
  float alpha = 1.f / (1.f + __expf(-diff));
  v4h hv = *(const v4h*)(h16 + (size_t)s * DCONST + lane * 4);
  float v0 = (float)hv.x, v1 = (float)hv.y, v2 = (float)hv.z, v3 = (float)hv.w;
  float* pa = agg_a + (size_t)d * DCONST + lane * 4;
  float* ps = agg_s + (size_t)d * DCONST + lane * 4;
  atomicAdd(pa + 0, v0 * alpha); atomicAdd(pa + 1, v1 * alpha);
  atomicAdd(pa + 2, v2 * alpha); atomicAdd(pa + 3, v3 * alpha);
  atomicAdd(ps + 0, v0);         atomicAdd(ps + 1, v1);
  atomicAdd(ps + 2, v2);         atomicAdd(ps + 3, v3);
}

// agg_a16 = f16(agg_a); agg_b16 = f16(agg_s - agg_a)   [inputs to fwd/bwd GEMMs]
__global__ void agg_cvt_kernel(const float* __restrict__ agg_a,
                               const float* __restrict__ agg_s,
                               _Float16* __restrict__ a16,
                               _Float16* __restrict__ b16, size_t n4) {
  size_t i = blockIdx.x * (size_t)blockDim.x + threadIdx.x;
  if (i >= n4) return;
  float4 a = *((const float4*)agg_a + i);
  float4 s = *((const float4*)agg_s + i);
  v4h oa = { (_Float16)a.x, (_Float16)a.y, (_Float16)a.z, (_Float16)a.w };
  v4h ob = { (_Float16)(s.x - a.x), (_Float16)(s.y - a.y),
             (_Float16)(s.z - a.z), (_Float16)(s.w - a.w) };
  *((v4h*)a16 + i) = oa;
  *((v4h*)b16 + i) = ob;
}

// ---- Helpers ---------------------------------------------------------------
__global__ void cvt_f32_to_f16(const float* __restrict__ in,
                               _Float16* __restrict__ out, size_t n4) {
  size_t i = blockIdx.x * (size_t)blockDim.x + threadIdx.x;
  if (i >= n4) return;
  float4 v = *((const float4*)in + i);
  v4h o = { (_Float16)v.x, (_Float16)v.y, (_Float16)v.z, (_Float16)v.w };
  *((v4h*)out + i) = o;
}

// Wscore [L,1,128] -> padded f16 [L,16,128] tiles (row 0 = Wscore, rest 0),
// so the score matvec rides the same WMMA path.
__global__ void pad_wscore_kernel(const float* __restrict__ Ws,
                                  _Float16* __restrict__ out, int L) {
  int i = blockIdx.x * blockDim.x + threadIdx.x;
  if (i >= L * 16 * DCONST) return;
  int c = i & (DCONST - 1);
  int r = (i >> 7) & 15;
  int l = i >> 11;
  float v = (r == 0) ? Ws[(size_t)l * DCONST + c] : 0.f;
  out[i] = (_Float16)v;
}

__global__ void zero_f32_kernel(float* __restrict__ p, size_t n) {
  size_t i = blockIdx.x * (size_t)blockDim.x + threadIdx.x;
  if (i < n) p[i] = 0.f;
}

__global__ void pool_accum_kernel(const float* __restrict__ h,
                                  const long long* __restrict__ batch,
                                  float* __restrict__ out,
                                  float* __restrict__ cnt, int N) {
  int n = blockIdx.x;
  int c = threadIdx.x;  // 128
  int g = (int)batch[n];
  atomicAdd(out + (size_t)g * DCONST + c, h[(size_t)n * DCONST + c]);
  if (c == 0) atomicAdd(cnt + g, 1.0f);
}

__global__ void pool_div_kernel(float* __restrict__ out,
                                const float* __restrict__ cnt, int total) {
  int i = blockIdx.x * blockDim.x + threadIdx.x;
  if (i < total) out[i] /= fmaxf(cnt[i >> 7], 1.0f);
}

// ---------------------------------------------------------------------------
static inline int cdiv(long long a, long long b) { return (int)((a + b - 1) / b); }

extern "C" void kernel_launch(void* const* d_in, const int* in_sizes, int n_in,
                              void* d_out, int out_size, void* d_ws, size_t ws_size,
                              hipStream_t stream) {
  const float*     x      = (const float*)d_in[0];
  const long long* edge   = (const long long*)d_in[1];
  const long long* batch  = (const long long*)d_in[2];
  const float*     Wemb   = (const float*)d_in[3];
  const float*     bemb   = (const float*)d_in[4];
  const float*     Wscore = (const float*)d_in[5];
  // d_in[6] = bscore: cancels inside sigmoid(s_i - s_j); intentionally unused.
  const float*     Wfwd   = (const float*)d_in[7];
  const float*     Wbwd   = (const float*)d_in[8];
  const float*     Wself  = (const float*)d_in[9];
  const float*     bself  = (const float*)d_in[10];
  const float*     Wcomb  = (const float*)d_in[11];
  const float*     bcomb  = (const float*)d_in[12];
  const float*     gamma  = (const float*)d_in[13];
  const float*     beta   = (const float*)d_in[14];

  const int D = DCONST;
  const int N = in_sizes[0] / D;            // 50000 (multiple of 16)
  const int E = in_sizes[1] / 2;            // 625000
  const int L = in_sizes[7] / (D * D);      // 3
  const int G = out_size / D;               // 64

  // ---- workspace carve-up (256B aligned slices) ----
  char* ws = (char*)d_ws;
  size_t off = 0;
  auto take = [&](size_t bytes) -> char* {
    char* p = ws + off;
    off = (off + bytes + 255) & ~(size_t)255;
    return p;
  };
  float*    hf32    = (float*)take((size_t)N * D * 4);
  _Float16* hf16    = (_Float16*)take((size_t)N * D * 2);
  _Float16* x16     = (_Float16*)take((size_t)N * D * 2);
  float*    agg_a   = (float*)take((size_t)N * D * 4);      // Σ α h_src
  float*    agg_s   = (float*)take((size_t)N * D * 4);      // Σ h_src
  _Float16* agg_a16 = (_Float16*)take((size_t)N * D * 2);
  _Float16* agg_b16 = (_Float16*)take((size_t)N * D * 2);
  float*    scores  = (float*)take((size_t)N * 16 * 4);     // [N,16], col 0 used
  _Float16* zc16    = (_Float16*)take((size_t)N * 3 * D * 2);
  _Float16* Wemb16  = (_Float16*)take((size_t)D * D * 2);
  _Float16* Wfwd16  = (_Float16*)take((size_t)L * D * D * 2);
  _Float16* Wbwd16  = (_Float16*)take((size_t)L * D * D * 2);
  _Float16* Wself16 = (_Float16*)take((size_t)L * D * D * 2);
  _Float16* Wcomb16 = (_Float16*)take((size_t)L * D * 3 * D * 2);
  _Float16* Wsc16   = (_Float16*)take((size_t)L * 16 * D * 2);
  float*    cnt     = (float*)take((size_t)G * 4);
  (void)ws_size;

  const int TB = 256;
  const int mtiles  = N / 16;
  const int gblocks = cdiv(mtiles, 8);      // 8 waves per block

  // ---- one-time conversions (re-done each call: deterministic) ----
  cvt_f32_to_f16<<<cdiv((long long)N * D / 4, TB), TB, 0, stream>>>(x, x16, (size_t)N * D / 4);
  cvt_f32_to_f16<<<cdiv((long long)D * D / 4, TB), TB, 0, stream>>>(Wemb, Wemb16, (size_t)D * D / 4);
  cvt_f32_to_f16<<<cdiv((long long)L * D * D / 4, TB), TB, 0, stream>>>(Wfwd, Wfwd16, (size_t)L * D * D / 4);
  cvt_f32_to_f16<<<cdiv((long long)L * D * D / 4, TB), TB, 0, stream>>>(Wbwd, Wbwd16, (size_t)L * D * D / 4);
  cvt_f32_to_f16<<<cdiv((long long)L * D * D / 4, TB), TB, 0, stream>>>(Wself, Wself16, (size_t)L * D * D / 4);
  cvt_f32_to_f16<<<cdiv((long long)L * D * 3 * D / 4, TB), TB, 0, stream>>>(Wcomb, Wcomb16, (size_t)L * D * 3 * D / 4);
  pad_wscore_kernel<<<cdiv(L * 16 * D, TB), TB, 0, stream>>>(Wscore, Wsc16, L);

  // ---- embedding: h = relu(x @ Wemb^T + bemb) ----
  gemm_f16_kernel<8, true><<<gblocks, TB, 0, stream>>>(
      x16, Wemb16, bemb, hf32, hf16, N, D, D, 0);

  // ---- layers ----
  for (int l = 0; l < L; ++l) {
    const _Float16* Wf  = Wfwd16  + (size_t)l * D * D;
    const _Float16* Wb  = Wbwd16  + (size_t)l * D * D;
    const _Float16* Wsf = Wself16 + (size_t)l * D * D;
    const _Float16* Wc  = Wcomb16 + (size_t)l * D * 3 * D;
    const _Float16* Wsc = Wsc16   + (size_t)l * 16 * D;

    // scores (16-wide padded WMMA tile; only col 0 meaningful)
    gemm_f16_kernel<1, false><<<gblocks, TB, 0, stream>>>(
        hf16, Wsc, nullptr, scores, nullptr, N, D, 16, 0);

    // gated aggregation of h itself (linearity: project aggregates afterwards)
    zero_f32_kernel<<<cdiv((long long)N * D, TB), TB, 0, stream>>>(agg_a, (size_t)N * D);
    zero_f32_kernel<<<cdiv((long long)N * D, TB), TB, 0, stream>>>(agg_s, (size_t)N * D);
    edge_aggr_kernel<<<cdiv((long long)E * 32, TB), TB, 0, stream>>>(
        edge, scores, hf16, agg_a, agg_s, E);
    agg_cvt_kernel<<<cdiv((long long)N * D / 4, TB), TB, 0, stream>>>(
        agg_a, agg_s, agg_a16, agg_b16, (size_t)N * D / 4);

    // projections write f16 straight into the [N,384] concat operand:
    //   cols [0,128)   = h @ Wself^T + bself
    //   cols [128,256) = (Σ α h) @ Wfwd^T
    //   cols [256,384) = (Σ (1-α) h) @ Wbwd^T
    gemm_f16_kernel<8, false><<<gblocks, TB, 0, stream>>>(
        hf16, Wsf, bself + (size_t)l * D, nullptr, zc16, N, D, 3 * D, 0);
    gemm_f16_kernel<8, false><<<gblocks, TB, 0, stream>>>(
        agg_a16, Wf, nullptr, nullptr, zc16, N, D, 3 * D, D);
    gemm_f16_kernel<8, false><<<gblocks, TB, 0, stream>>>(
        agg_b16, Wb, nullptr, nullptr, zc16, N, D, 3 * D, 2 * D);

    // comb GEMM + LN + relu + residual (in-place h update)
    comb_ln_kernel<<<gblocks, TB, 0, stream>>>(
        zc16, Wc, bcomb + (size_t)l * D,
        gamma + (size_t)l * D, beta + (size_t)l * D, hf32, hf16, N);
  }

  // ---- global mean pool per graph ----
  zero_f32_kernel<<<cdiv(out_size, TB), TB, 0, stream>>>((float*)d_out, (size_t)out_size);
  zero_f32_kernel<<<1, TB, 0, stream>>>(cnt, (size_t)G);
  pool_accum_kernel<<<N, D, 0, stream>>>(hf32, batch, (float*)d_out, cnt, N);
  pool_div_kernel<<<cdiv(out_size, TB), TB, 0, stream>>>((float*)d_out, cnt, out_size);
}
